// SimpleRNN_33294586479163
// MI455X (gfx1250) — matrix operand order repaired and checked by reading, not compiled, not run on hardware
//
#include <hip/hip_runtime.h>
#include <hip/hip_bf16.h>
#include <math.h>

typedef __attribute__((ext_vector_type(16))) _Float16 v16h;
typedef __attribute__((ext_vector_type(8)))  _Float16 v8h;
typedef __attribute__((ext_vector_type(8)))  float    v8f;

#define B_   128
#define T_   1024
#define IN_  256
#define H_   512
#define OUT_ 256
#define NWG  16
#define NTHR 256

// ---------------- fragment loaders ----------------
// A (f16 16x32, row-major src): lane L -> row m0+(L&15),
// 16 contiguous halves starting at k0+(L>>4)*16.
__device__ __forceinline__ v16h load_a_frag(const _Float16* __restrict__ base, int ld, int lane) {
  const _Float16* p = base + (size_t)(lane & 15) * ld + ((lane >> 4) << 4);
  union { v16h v; v8h h[2]; } u;
  u.h[0] = *(const v8h*)(p);
  u.h[1] = *(const v8h*)(p + 8);
  return u.v;
}
// B (f16 32x16) from row-major W[n][k] (B = W^T): lane L -> row n0+(L&15),
// 16 contiguous halves starting at k0+(L>>4)*16.
__device__ __forceinline__ v16h load_b_frag(const _Float16* base, int ld, int lane) {
  return *(const v16h*)(base + (size_t)(lane & 15) * ld + ((lane >> 4) << 4));
}

__device__ __forceinline__ v8f wmma_f16(v16h a, v16h b, v8f c) {
  return __builtin_amdgcn_wmma_f32_16x16x32_f16(false, a, false, b, (short)0, c, false, false);
}

__device__ __forceinline__ v8f cvt_v8h_to_v8f(v8h h) {
  v8f f;
  #pragma unroll
  for (int r = 0; r < 8; ++r) f[r] = (float)h[r];
  return f;
}

// ---------------- grid barrier (monotonic counter, re-zeroed per launch) ----------------
__device__ __forceinline__ void grid_sync(unsigned* cnt, unsigned target) {
  __threadfence();
  __syncthreads();
  if (threadIdx.x == 0) {
    __hip_atomic_fetch_add(cnt, 1u, __ATOMIC_ACQ_REL, __HIP_MEMORY_SCOPE_AGENT);
    while (__hip_atomic_load(cnt, __ATOMIC_ACQUIRE, __HIP_MEMORY_SCOPE_AGENT) < target) {
      __builtin_amdgcn_s_sleep(2);
    }
  }
  __syncthreads();
  __threadfence();
}

// ---------------- prep: weights f32->f16, bias sums, h-init, counter reset ----------------
__global__ void prep_misc_kernel(const float* __restrict__ w_ih0, const float* __restrict__ w_hh0,
                                 const float* __restrict__ b_ih0, const float* __restrict__ b_hh0,
                                 const float* __restrict__ w_ih1, const float* __restrict__ w_hh1,
                                 const float* __restrict__ b_ih1, const float* __restrict__ b_hh1,
                                 const float* __restrict__ fc_w,  const float* __restrict__ h0in,
                                 _Float16* __restrict__ wih0h, _Float16* __restrict__ whh0h,
                                 _Float16* __restrict__ wih1h, _Float16* __restrict__ whh1h,
                                 _Float16* __restrict__ fcwh,
                                 float* __restrict__ bias0, float* __restrict__ bias1,
                                 _Float16* __restrict__ h0b, _Float16* __restrict__ h1b,
                                 unsigned* __restrict__ cnt) {
  const size_t i = (size_t)blockIdx.x * blockDim.x + threadIdx.x;
  if (i == 0) cnt[0] = 0u;
  if (i < (size_t)H_ * IN_)  wih0h[i] = (_Float16)w_ih0[i];
  if (i < (size_t)H_ * H_) {
    whh0h[i] = (_Float16)w_hh0[i];
    wih1h[i] = (_Float16)w_ih1[i];
    whh1h[i] = (_Float16)w_hh1[i];
  }
  if (i < (size_t)OUT_ * H_) fcwh[i] = (_Float16)fc_w[i];
  if (i < (size_t)H_) { bias0[i] = b_ih0[i] + b_hh0[i]; bias1[i] = b_ih1[i] + b_hh1[i]; }
  if (i < (size_t)B_ * H_) {
    h0b[i] = (_Float16)h0in[i];
    h1b[i] = (_Float16)h0in[(size_t)B_ * H_ + i];
  }
}

// ---------------- prep: x [B,T,IN] f32 -> xT [T,B,IN] f16 ----------------
__global__ void prep_x_kernel(const float* __restrict__ x, _Float16* __restrict__ xT) {
  const size_t i = (size_t)blockIdx.x * blockDim.x + threadIdx.x;
  if (i < (size_t)B_ * T_ * IN_) {
    const int c = (int)(i % IN_);
    const int t = (int)((i / IN_) % T_);
    const int b = (int)(i / ((size_t)T_ * IN_));
    xT[(size_t)t * B_ * IN_ + (size_t)b * IN_ + c] = (_Float16)x[i];
  }
}

// ---------------- parallel pre-GEMM: X0 = x_t @ wih0^T + bias0, all t ----------------
// Output layout: C-fragment-permuted f16: X0p[tile][lane][8], tile = (t*8+mi)*32+nj.
// One 16x16 tile per wave; 262144 tiles total.
__global__ __launch_bounds__(NTHR)
void x_gemm_kernel(const _Float16* __restrict__ xT, const _Float16* __restrict__ wih0,
                   const float* __restrict__ bias0, _Float16* __restrict__ X0p) {
  const int lane = threadIdx.x & 31;
  const int wave = threadIdx.x >> 5;
  const size_t tile = (size_t)blockIdx.x * 8 + wave;   // 0 .. T_*8*32-1
  const int nj = (int)(tile & 31);
  const int mi = (int)((tile >> 5) & 7);
  const int t  = (int)(tile >> 8);
  const _Float16* Ax = xT + (size_t)t * (B_ * IN_) + (size_t)mi * 16 * IN_;
  const _Float16* Bw = wih0 + (size_t)nj * 16 * IN_;
  v8f acc = {0.f, 0.f, 0.f, 0.f, 0.f, 0.f, 0.f, 0.f};
  #pragma unroll
  for (int k0 = 0; k0 < IN_; k0 += 32) {
    acc = wmma_f16(load_a_frag(Ax + k0, IN_, lane), load_b_frag(Bw + k0, IN_, lane), acc);
  }
  const float bv = bias0[nj * 16 + (lane & 15)];
  v8h o;
  #pragma unroll
  for (int r = 0; r < 8; ++r) o[r] = (_Float16)(acc[r] + bv);
  *(v8h*)(X0p + tile * 256 + (size_t)lane * 8) = o;
}

// ---------------- persistent 2-layer RNN scan ----------------
__global__ __launch_bounds__(NTHR, 1)
void rnn_scan_kernel(const _Float16* __restrict__ X0p,  // precomputed x@wih0^T + bias0
                     _Float16* __restrict__ h0buf,      // [2][B][H] f16
                     _Float16* __restrict__ h1buf,      // [2][B][H] f16
                     const _Float16* __restrict__ whh0,
                     const _Float16* __restrict__ wih1, const _Float16* __restrict__ whh1,
                     const float* __restrict__ bias1,
                     unsigned* __restrict__ cnt) {
  extern __shared__ _Float16 lds[];
  _Float16* s_whh0 = lds;                   // 32 x H_
  _Float16* s_wih1 = s_whh0 + 32 * H_;      // 32 x H_
  _Float16* s_whh1 = s_wih1 + 32 * H_;      // 32 x H_

  const int tid  = threadIdx.x;
  const int lane = tid & 31;
  const int wave = tid >> 5;                // 0..7 -> M tile
  const int n0g  = blockIdx.x * 32;         // this WG owns 32 output columns

  // Load this WG's weight-row slices into LDS once (rows n0g..n0g+31 are contiguous).
  {
    const uint4* s1 = (const uint4*)(whh0 + (size_t)n0g * H_);
    const uint4* s2 = (const uint4*)(wih1 + (size_t)n0g * H_);
    const uint4* s3 = (const uint4*)(whh1 + (size_t)n0g * H_);
    uint4* d1 = (uint4*)s_whh0; uint4* d2 = (uint4*)s_wih1; uint4* d3 = (uint4*)s_whh1;
    for (int i = tid; i < 32 * H_ / 8; i += NTHR) { d1[i] = s1[i]; d2[i] = s2[i]; d3[i] = s3[i]; }
  }
  __syncthreads();

  const int ncl  = lane & 15;
  const int mrow = wave * 16 + ((lane >> 4) << 3);   // C/D layout row base
  const int mi   = wave;
  const int njg0 = (n0g >> 4);                       // global N-tile index of nloc=0
  // per-wave X0p offsets (halves) for its two tiles at timestep t:
  //   ((t*8 + mi)*32 + njg) * 256 + lane*8
  const size_t xoff0 = ((size_t)mi * 32 + njg0) * 256 + (size_t)lane * 8;
  const size_t xoff1 = xoff0 + 256;

  unsigned epoch = 0;
  int p = 0;
  for (int t = 0; t < T_; ++t) {
    const size_t tbase = (size_t)t * (8 * 32 * 256);
    // -------- layer 0: h0n = tanh(X0[t] + h0 @ whh0^T) --------
    const _Float16* Ah  = h0buf + (size_t)p * (B_ * H_);
    _Float16*       H0n = h0buf + (size_t)(1 - p) * (B_ * H_);

    if (t + 1 < T_) {   // warm next step's precomputed tiles (global_prefetch_b8)
      __builtin_prefetch(X0p + tbase + (size_t)(8 * 32 * 256) + xoff0, 0, 1);
      __builtin_prefetch(X0p + tbase + (size_t)(8 * 32 * 256) + xoff1, 0, 1);
    }

    {
      v8f acc0 = cvt_v8h_to_v8f(*(const v8h*)(X0p + tbase + xoff0));
      v8f acc1 = cvt_v8h_to_v8f(*(const v8h*)(X0p + tbase + xoff1));
      #pragma unroll
      for (int k0 = 0; k0 < H_; k0 += 32) {
        v16h a  = load_a_frag(Ah + (size_t)mi * 16 * H_ + k0, H_, lane);
        v16h b0 = load_b_frag(s_whh0 + k0, H_, lane);
        v16h b1 = load_b_frag(s_whh0 + (size_t)16 * H_ + k0, H_, lane);
        acc0 = wmma_f16(a, b0, acc0);
        acc1 = wmma_f16(a, b1, acc1);
      }
      const int n0 = n0g + ncl;
      const int n1 = n0g + 16 + ncl;
      #pragma unroll
      for (int r = 0; r < 8; ++r) {
        H0n[(size_t)(mrow + r) * H_ + n0] = (_Float16)tanhf(acc0[r]);
        H0n[(size_t)(mrow + r) * H_ + n1] = (_Float16)tanhf(acc1[r]);
      }
    }
    ++epoch; grid_sync(cnt, epoch * NWG);

    // -------- layer 1: h1n = tanh(h0n @ wih1^T + h1 @ whh1^T + bias1) --------
    const _Float16* A0  = h0buf + (size_t)(1 - p) * (B_ * H_);
    const _Float16* A1  = h1buf + (size_t)p * (B_ * H_);
    _Float16*       H1n = h1buf + (size_t)(1 - p) * (B_ * H_);
    {
      v8f acc0 = {0.f, 0.f, 0.f, 0.f, 0.f, 0.f, 0.f, 0.f};
      v8f acc1 = {0.f, 0.f, 0.f, 0.f, 0.f, 0.f, 0.f, 0.f};
      #pragma unroll
      for (int k0 = 0; k0 < H_; k0 += 32) {
        v16h a  = load_a_frag(A0 + (size_t)mi * 16 * H_ + k0, H_, lane);
        v16h b0 = load_b_frag(s_wih1 + k0, H_, lane);
        v16h b1 = load_b_frag(s_wih1 + (size_t)16 * H_ + k0, H_, lane);
        acc0 = wmma_f16(a, b0, acc0);
        acc1 = wmma_f16(a, b1, acc1);
      }
      #pragma unroll
      for (int k0 = 0; k0 < H_; k0 += 32) {
        v16h a  = load_a_frag(A1 + (size_t)mi * 16 * H_ + k0, H_, lane);
        v16h b0 = load_b_frag(s_whh1 + k0, H_, lane);
        v16h b1 = load_b_frag(s_whh1 + (size_t)16 * H_ + k0, H_, lane);
        acc0 = wmma_f16(a, b0, acc0);
        acc1 = wmma_f16(a, b1, acc1);
      }
      const int n0 = n0g + ncl;
      const int n1 = n0g + 16 + ncl;
      const float bv0 = bias1[n0];
      const float bv1 = bias1[n1];
      #pragma unroll
      for (int r = 0; r < 8; ++r) {
        H1n[(size_t)(mrow + r) * H_ + n0] = (_Float16)tanhf(acc0[r] + bv0);
        H1n[(size_t)(mrow + r) * H_ + n1] = (_Float16)tanhf(acc1[r] + bv1);
      }
    }
    ++epoch; grid_sync(cnt, epoch * NWG);
    p ^= 1;
  }
}

// ---------------- final FC: out = h1_T @ fc_w^T + fc_b  (f32 out) ----------------
__global__ __launch_bounds__(NTHR)
void fc_kernel(const _Float16* __restrict__ h1, const _Float16* __restrict__ fcw,
               const float* __restrict__ fcb, float* __restrict__ out) {
  const int lane = threadIdx.x & 31;
  const int wave = threadIdx.x >> 5;
  const int gid  = blockIdx.x * 8 + wave;   // 0..127 tiles
  const int mi   = gid >> 4;                // 0..7
  const int ni   = gid & 15;                // 0..15
  v8f acc = {0.f, 0.f, 0.f, 0.f, 0.f, 0.f, 0.f, 0.f};
  #pragma unroll
  for (int k0 = 0; k0 < H_; k0 += 32) {
    v16h a = load_a_frag(h1 + (size_t)mi * 16 * H_ + k0, H_, lane);
    v16h b = load_b_frag(fcw + (size_t)ni * 16 * H_ + k0, H_, lane);
    acc = wmma_f16(a, b, acc);
  }
  const int n = ni * 16 + (lane & 15);
  const float bv = fcb[n];
  const int mrow = mi * 16 + ((lane >> 4) << 3);
  #pragma unroll
  for (int r = 0; r < 8; ++r) {
    out[(size_t)(mrow + r) * OUT_ + n] = acc[r] + bv;
  }
}

// ---------------- host ----------------
extern "C" void kernel_launch(void* const* d_in, const int* in_sizes, int n_in,
                              void* d_out, int out_size, void* d_ws, size_t ws_size,
                              hipStream_t stream) {
  (void)in_sizes; (void)n_in; (void)out_size; (void)ws_size;
  const float* x     = (const float*)d_in[0];
  const float* h0in  = (const float*)d_in[1];
  const float* w_ih0 = (const float*)d_in[2];
  const float* w_hh0 = (const float*)d_in[3];
  const float* b_ih0 = (const float*)d_in[4];
  const float* b_hh0 = (const float*)d_in[5];
  const float* w_ih1 = (const float*)d_in[6];
  const float* w_hh1 = (const float*)d_in[7];
  const float* b_ih1 = (const float*)d_in[8];
  const float* b_hh1 = (const float*)d_in[9];
  const float* fc_w  = (const float*)d_in[10];
  const float* fc_b  = (const float*)d_in[11];
  float* out = (float*)d_out;

  char* ws = (char*)d_ws;
  size_t cur = 0;
  auto carve = [&](size_t bytes) -> char* {
    char* p = ws + cur;
    cur += (bytes + 255) & ~(size_t)255;
    return p;
  };
  unsigned* sync_cnt = (unsigned*)carve(256);
  _Float16* xT    = (_Float16*)carve((size_t)T_ * B_ * IN_ * 2);
  _Float16* X0p   = (_Float16*)carve((size_t)T_ * B_ * H_ * 2);   // permuted x@wih0^T+bias0
  _Float16* h0b   = (_Float16*)carve((size_t)2 * B_ * H_ * 2);
  _Float16* h1b   = (_Float16*)carve((size_t)2 * B_ * H_ * 2);
  _Float16* wih0h = (_Float16*)carve((size_t)H_ * IN_ * 2);
  _Float16* whh0h = (_Float16*)carve((size_t)H_ * H_ * 2);
  _Float16* wih1h = (_Float16*)carve((size_t)H_ * H_ * 2);
  _Float16* whh1h = (_Float16*)carve((size_t)H_ * H_ * 2);
  _Float16* fcwh  = (_Float16*)carve((size_t)OUT_ * H_ * 2);
  float* bias0    = (float*)carve(H_ * 4);
  float* bias1    = (float*)carve(H_ * 4);

  // 1) convert weights/biases/h-init, reset grid barrier counter
  prep_misc_kernel<<<(H_ * H_ + NTHR - 1) / NTHR, NTHR, 0, stream>>>(
      w_ih0, w_hh0, b_ih0, b_hh0, w_ih1, w_hh1, b_ih1, b_hh1, fc_w, h0in,
      wih0h, whh0h, wih1h, whh1h, fcwh, bias0, bias1, h0b, h1b, sync_cnt);
  // 2) x -> f16, [T,B,IN]
  const size_t nx = (size_t)B_ * T_ * IN_;
  prep_x_kernel<<<(unsigned)((nx + NTHR - 1) / NTHR), NTHR, 0, stream>>>(x, xT);
  // 3) parallel pre-GEMM of the input term for all timesteps (off the critical path)
  const unsigned ntiles = T_ * 8 * 32;
  x_gemm_kernel<<<ntiles / 8, NTHR, 0, stream>>>(xT, wih0h, bias0, X0p);
  // 4) persistent scan (recurrent weights resident in 96KB LDS per WG)
  const size_t ldsBytes = (size_t)(3 * 32 * H_) * sizeof(_Float16);
  rnn_scan_kernel<<<NWG, NTHR, ldsBytes, stream>>>(X0p, h0b, h1b, whh0h,
                                                   wih1h, whh1h, bias1, sync_cnt);
  // 5) final FC; T_ is even so final h1 is in parity-0 buffer (h1b base)
  fc_kernel<<<NWG, NTHR, 0, stream>>>(h1b, fcwh, fc_b, out);
}
